// MultiheadAttention_15633680957669
// MI455X (gfx1250) — matrix operand order repaired and checked by reading, not compiled, not run on hardware
//
#include <hip/hip_runtime.h>

#define B_    4
#define S_    2048
#define D_    1024
#define H_    16
#define HD_   64
#define QKVN  3072

typedef _Float16 h16;
typedef __attribute__((ext_vector_type(16))) _Float16 v16h;
typedef __attribute__((ext_vector_type(8)))  _Float16 v8h;
typedef __attribute__((ext_vector_type(4)))  _Float16 v4h;
typedef __attribute__((ext_vector_type(8)))  float    v8f;
typedef __attribute__((ext_vector_type(4)))  float    v4f;

// log2(e) / sqrt(HD): folded into Q so softmax runs in exp2 domain
#define QSCALE 0.18033688f

static __device__ __forceinline__ v8f wmma_f16(v16h a, v16h b, v8f c) {
  return __builtin_amdgcn_wmma_f32_16x16x32_f16(
      /*neg_a=*/false, a, /*neg_b=*/false, b,
      /*c_mod=*/(short)0, c, /*reuse_a=*/false, /*reuse_b=*/false);
}

// A-fragment (contiguous k order k = 16*hi + s at every call): per lane two contiguous 8-half chunks.
static __device__ __forceinline__ v16h frag2(const h16* p0, const h16* p1) {
  v8h a = *(const v8h*)p0;
  v8h b = *(const v8h*)p1;
  v16h r;
#pragma unroll
  for (int i = 0; i < 8; ++i) { r[i] = a[i]; r[i + 8] = b[i]; }
  return r;
}

// ---------------------------------------------------------------- converts (x4 vectorized)
__global__ void cvt_f32_f16(const float* __restrict__ in, h16* __restrict__ out, long n4) {
  long i = (long)blockIdx.x * blockDim.x + threadIdx.x;
  long stride = (long)gridDim.x * blockDim.x;
  for (; i < n4; i += stride) {
    v4f x = ((const v4f*)in)[i];
    v4h o;
#pragma unroll
    for (int j = 0; j < 4; ++j) o[j] = (h16)x[j];
    ((v4h*)out)[i] = o;
  }
}

// ---------------------------------------------------------------- QKV GEMM
// X[8192,1024]f16 * W[3072,1024]^T -> Q[b,h,s,64]*QSCALE, K[b,h,s,64], Vt[b,h,64,s]
// wave tile: 32 rows x 64 cols (2 row-tiles x 4 col-tiles), 8 WMMA / k-step
__global__ void __launch_bounds__(256) qkv_gemm(const h16* __restrict__ X,
                                                const h16* __restrict__ W,
                                                h16* __restrict__ Q,
                                                h16* __restrict__ K,
                                                h16* __restrict__ Vt) {
  const int lane = threadIdx.x & 31;
  const int wv   = threadIdx.x >> 5;
  const int gw   = blockIdx.x * 8 + wv;
  const int NT   = QKVN / 64;            // 48 col-tiles
  const int mt   = gw / NT;              // 32-row tile index (256 total)
  const int nt   = gw % NT;
  const int n = lane & 15, hi = lane >> 4;

  const h16* xrow0 = X + ((long)mt * 32 + n)      * D_;
  const h16* xrow1 = X + ((long)mt * 32 + 16 + n) * D_;
  v8f acc[2][4] = {};

  for (int k = 0; k < D_; k += 32) {
    v16h a0 = frag2(xrow0 + k + (hi ? 16 : 0), xrow0 + k + 8 + (hi ? 16 : 0));
    v16h a1 = frag2(xrow1 + k + (hi ? 16 : 0), xrow1 + k + 8 + (hi ? 16 : 0));
#pragma unroll
    for (int tt = 0; tt < 4; ++tt) {
      const int j0 = nt * 64 + tt * 16;
      v16h b = *(const v16h*)(W + (long)(j0 + n) * D_ + k + (hi ? 16 : 0));
      acc[0][tt] = wmma_f16(a0, b, acc[0][tt]);
      acc[1][tt] = wmma_f16(a1, b, acc[1][tt]);
    }
  }

#pragma unroll
  for (int rt = 0; rt < 2; ++rt) {
    const int g0   = mt * 32 + rt * 16;  // 32-row tile never straddles a batch
    const int bidx = g0 / S_;
    const int srow = g0 % S_;
#pragma unroll
    for (int tt = 0; tt < 4; ++tt) {
      const int j0   = nt * 64 + tt * 16;
      const int h    = j0 / 192;         // 192 = 3*head_dim channel group
      const int c    = j0 % 192;
      const int part = c / 64;           // 0=q 1=k 2=v
      const int c0   = c % 64;
      if (part == 2) {
        v8h t;
#pragma unroll
        for (int r = 0; r < 8; ++r) t[r] = (h16)acc[rt][tt][r];
        h16* dst = Vt + ((long)(bidx * H_ + h) * HD_ + c0 + n) * S_ + srow + 8 * hi;
        *(v8h*)dst = t;                  // 8 consecutive s -> one 16B store
      } else {
        h16* base = (part == 0 ? Q : K) + (long)(bidx * H_ + h) * S_ * HD_;
        const float sc = (part == 0) ? QSCALE : 1.0f;
#pragma unroll
        for (int r = 0; r < 8; ++r) {
          const int s = srow + r + 8 * hi;
          base[(long)s * HD_ + c0 + n] = (h16)(acc[rt][tt][r] * sc);
        }
      }
    }
  }
}

// ---------------------------------------------------------------- flash attention
// one wave per 16-row q tile; online softmax (exp2 domain) over kj tiles of 64
__global__ void __launch_bounds__(32) attn_fwd(const h16* __restrict__ Q,
                                               const h16* __restrict__ K,
                                               const h16* __restrict__ Vt,
                                               h16* __restrict__ Ov) {
  __shared__ h16 plds[16][72];           // 16x64 P tile, rows padded to 144B
  const int lane = threadIdx.x & 31;
  const int n = lane & 15, hi = lane >> 4;

  const int t  = blockIdx.x;
  const int qt = t & 127;                // S_/16 = 128
  const int bh = t >> 7;
  const int b  = bh >> 4, h = bh & 15;

  const h16* Qb = Q  + (long)(b * H_ + h) * S_ * HD_;
  const h16* Kb = K  + (long)(b * H_ + h) * S_ * HD_;
  const h16* Vb = Vt + (long)(b * H_ + h) * HD_ * S_;
  const int sq = qt * 16;

  const h16* qrow = Qb + (long)(sq + n) * HD_;
  const v16h aq0 = frag2(qrow +      (hi ? 16 : 0), qrow + 8 + (hi ? 16 : 0));
  const v16h aq1 = frag2(qrow + 32 + (hi ? 16 : 0), qrow + 40 + (hi ? 16 : 0));

  float rowmax[8], rowsum[8], corr[8];
#pragma unroll
  for (int r = 0; r < 8; ++r) { rowmax[r] = -3.0e38f; rowsum[r] = 0.f; }
  v8f acc[4] = {};                       // O tile 16x64

  for (int kj = 0; kj < S_; kj += 64) {
    // ---- logits: 4 col-fragments x (d=0..31, 32..63)
    v8f s[4];
#pragma unroll
    for (int cf = 0; cf < 4; ++cf) {
      const h16* kr = Kb + (long)(kj + cf * 16 + n) * HD_ + (hi ? 16 : 0);
      v8f sv = {};
      sv = wmma_f16(aq0, *(const v16h*)(kr),      sv);
      sv = wmma_f16(aq1, *(const v16h*)(kr + 32), sv);
      s[cf] = sv;
    }

    // ---- online softmax per row m = r + 8*hi (C/D layout rows)
#pragma unroll
    for (int r = 0; r < 8; ++r) {
      const float l0 = s[0][r], l1 = s[1][r], l2 = s[2][r], l3 = s[3][r];
      float mx = fmaxf(fmaxf(l0, l1), fmaxf(l2, l3));
#pragma unroll
      for (int m = 1; m < 16; m <<= 1) mx = fmaxf(mx, __shfl_xor(mx, m, 32));
      const float nm = fmaxf(rowmax[r], mx);
      const float cr = exp2f(rowmax[r] - nm);
      const float p0 = exp2f(l0 - nm);
      const float p1 = exp2f(l1 - nm);
      const float p2 = exp2f(l2 - nm);
      const float p3 = exp2f(l3 - nm);
      float ps = (p0 + p1) + (p2 + p3);
#pragma unroll
      for (int m = 1; m < 16; m <<= 1) ps += __shfl_xor(ps, m, 32);
      rowsum[r] = rowsum[r] * cr + ps;
      rowmax[r] = nm;
      corr[r]   = cr;
      plds[r + 8 * hi][n]      = (h16)p0;
      plds[r + 8 * hi][16 + n] = (h16)p1;
      plds[r + 8 * hi][32 + n] = (h16)p2;
      plds[r + 8 * hi][48 + n] = (h16)p3;
    }
#pragma unroll
    for (int f = 0; f < 4; ++f)
#pragma unroll
      for (int r = 0; r < 8; ++r) acc[f][r] *= corr[r];

    __syncthreads();
    // ---- P (D-layout) -> A-layout via LDS; O += P V  (two 32-k steps)
    const h16* pr = &plds[n][0];
    const v16h ap0 = frag2(pr +      (hi ? 16 : 0), pr + 8 + (hi ? 16 : 0));
    const v16h ap1 = frag2(pr + 32 + (hi ? 16 : 0), pr + 40 + (hi ? 16 : 0));
#pragma unroll
    for (int f = 0; f < 4; ++f) {
      const h16* vp = Vb + (long)(f * 16 + n) * S_ + kj + (hi ? 16 : 0);
      acc[f] = wmma_f16(ap0, *(const v16h*)(vp),      acc[f]);
      acc[f] = wmma_f16(ap1, *(const v16h*)(vp + 32), acc[f]);
    }
    __syncthreads();
  }

  float inv[8];
#pragma unroll
  for (int r = 0; r < 8; ++r) inv[r] = 1.0f / rowsum[r];
  h16* orow = Ov + ((long)(b * S_ + sq)) * D_ + h * HD_;
#pragma unroll
  for (int f = 0; f < 4; ++f)
#pragma unroll
    for (int r = 0; r < 8; ++r) {
      const int m = r + 8 * hi;
      orow[(long)m * D_ + f * 16 + n] = (h16)(acc[f][r] * inv[r]);
    }
}

// ---------------------------------------------------------------- output GEMM
// vals[8192,1024]f16 * o_w[1024,1024]^T -> out fp32; wave tile 32x64
__global__ void __launch_bounds__(256) out_gemm(const h16* __restrict__ Vl,
                                                const h16* __restrict__ Wo,
                                                float* __restrict__ out) {
  const int lane = threadIdx.x & 31;
  const int wv   = threadIdx.x >> 5;
  const int gw   = blockIdx.x * 8 + wv;
  const int NT   = D_ / 64;              // 16 col-tiles
  const int mt   = gw / NT;              // 32-row tile (256 total)
  const int nt   = gw % NT;
  const int n = lane & 15, hi = lane >> 4;

  const h16* xrow0 = Vl + ((long)mt * 32 + n)      * D_;
  const h16* xrow1 = Vl + ((long)mt * 32 + 16 + n) * D_;
  v8f acc[2][4] = {};

  for (int k = 0; k < D_; k += 32) {
    v16h a0 = frag2(xrow0 + k + (hi ? 16 : 0), xrow0 + k + 8 + (hi ? 16 : 0));
    v16h a1 = frag2(xrow1 + k + (hi ? 16 : 0), xrow1 + k + 8 + (hi ? 16 : 0));
#pragma unroll
    for (int tt = 0; tt < 4; ++tt) {
      const int j0 = nt * 64 + tt * 16;
      v16h b = *(const v16h*)(Wo + (long)(j0 + n) * D_ + k + (hi ? 16 : 0));
      acc[0][tt] = wmma_f16(a0, b, acc[0][tt]);
      acc[1][tt] = wmma_f16(a1, b, acc[1][tt]);
    }
  }
#pragma unroll
  for (int rt = 0; rt < 2; ++rt) {
    const long rbase = (long)mt * 32 + rt * 16;
#pragma unroll
    for (int tt = 0; tt < 4; ++tt) {
      const int j0 = nt * 64 + tt * 16;
#pragma unroll
      for (int r = 0; r < 8; ++r) {
        const long g = rbase + r + 8 * hi;
        out[g * D_ + j0 + n] = acc[rt][tt][r];
      }
    }
  }
}

// ---------------------------------------------------------------- launcher
extern "C" void kernel_launch(void* const* d_in, const int* in_sizes, int n_in,
                              void* d_out, int out_size, void* d_ws, size_t ws_size,
                              hipStream_t stream) {
  (void)in_sizes; (void)n_in; (void)out_size; (void)ws_size;
  const float* x    = (const float*)d_in[0];
  const float* qkvw = (const float*)d_in[1];
  const float* ow   = (const float*)d_in[2];
  float* out = (float*)d_out;

  const long NX = (long)B_ * S_ * D_;       // 8,388,608
  const long NW = (long)QKVN * D_;          // 3,145,728
  const long NO = (long)D_ * D_;            // 1,048,576
  const long NQ = (long)B_ * H_ * S_ * HD_; // 8,388,608

  h16* p  = (h16*)d_ws;
  h16* xh = p; p += NX;
  h16* wh = p; p += NW;
  h16* oh = p; p += NO;
  h16* Qh = p; p += NQ;
  h16* Kh = p; p += NQ;
  h16* Vt = p; p += NQ;
  h16* vl = p; p += NX;

  cvt_f32_f16<<<1024, 256, 0, stream>>>(x,    xh, NX / 4);
  cvt_f32_f16<<< 512, 256, 0, stream>>>(qkvw, wh, NW / 4);
  cvt_f32_f16<<< 256, 256, 0, stream>>>(ow,   oh, NO / 4);

  // 256 row-tiles(32) * 48 col-tiles(64) = 12288 waves / 8 per block
  qkv_gemm<<<1536, 256, 0, stream>>>(xh, wh, Qh, Kh, Vt);

  // B*H*(S/16) = 8192 q-tiles, one wave each
  attn_fwd<<<8192, 32, 0, stream>>>(Qh, Kh, Vt, vl);

  // 256 row-tiles(32) * 16 col-tiles(64) = 4096 waves / 8 per block
  out_gemm<<<512, 256, 0, stream>>>(vl, oh, out);
}
